// RigPropagationModel_89721866813785
// MI455X (gfx1250) — matrix operand order repaired and checked by reading, not compiled, not run on hardware
//
#include <hip/hip_runtime.h>
#include <math.h>

// ---------------- types ----------------
typedef _Float16 h16 __attribute__((ext_vector_type(16)));
typedef _Float16 h8v __attribute__((ext_vector_type(8)));
typedef float    f8v __attribute__((ext_vector_type(8)));

// ---------------- problem constants ----------------
#define BATCH  4096
#define JNT    22
#define INF    10
#define NDIM   128
#define EDIM   32
#define FFND   2048
#define NLAY   4
#define NEDGE  42
#define TBATCH 8
#define TROWS  176      // TBATCH*JNT, 11 tiles of 16
#define TEDGES 336      // TBATCH*NEDGE, 21 tiles of 16
#define NWAVE  11
#define TPB    352      // 11 waves * 32

// ---------------- packed f16 weight layout (halves), per layer ----------------
#define EW1_O 0               // edge_w1 288x128
#define EW2_O 36864           // edge_w2 128x32
#define MW_O  40960           // msg_w   32x128
#define RW_O  45056           // r_w     256x128
#define UW_O  77824           // u_w     256x128
#define CW_O  110592          // c_w     256x128
#define FW1_O 143360          // ffn_w1  128x2048
#define FW2_O 405504          // ffn_w2  2048x128
#define LAYER_H 667648
#define TOT_H   (4L*LAYER_H)

// ---------------- dynamic LDS layout (bytes) ----------------
#define OFF_NORM 0            // 176*128 f16 = 45056
#define OFF_PROJ 45056        // 176*128 f16
#define OFF_R    90112        // 176*128 f16
#define OFF_EA   135168       // 2 * 336*32 f16 = 43008 (ping-pong edge_attr/msg)
#define OFF_AGG  178176       // 176*32 f16 = 11264 (reused as f32 raw[176][4] for head)
#define OFF_SCR  189440       // 11 waves * 16*128 f16 = 45056
#define OFF_IDX  234496       // src[42], tgt[42], edir[42] int
#define SMEM_BYTES 235008

// ---------------- helpers ----------------
static __device__ __forceinline__ f8v wmma_f16(h16 a, h16 b, f8v c) {
  return __builtin_amdgcn_wmma_f32_16x16x32_f16(false, a, false, b, (short)0, c, false, false);
}

// A fragment (16x32 f16) from a row-major f16 row chunk of 32 halves.
// Per ISA layout: lane l holds row m=l&15; element e -> k=(e>>3)*16+(e&7)+8*(l>>4).
static __device__ __forceinline__ h16 load_a(const _Float16* cb, int half) {
  h8v lo = *(const h8v*)(cb + 8 * half);
  h8v hi = *(const h8v*)(cb + 16 + 8 * half);
  return __builtin_shufflevector(lo, hi, 0,1,2,3,4,5,6,7,8,9,10,11,12,13,14,15);
}

// B fragment (32x16 f16) from row-major f16 weight W[K][N].
// Per ISA layout: lane l holds n = 16*nt + (l&15); element e -> k = 32*kt + (e>>3)*16 + (e&7) + 8*(l>>4)
static __device__ __forceinline__ h16 load_b(const _Float16* W, int N, int kt, int nt, int lane) {
  const _Float16* p = W + (size_t)(kt * 32 + 8 * (lane >> 4)) * N + nt * 16 + (lane & 15);
  h16 r;
#pragma unroll
  for (int s = 0; s < 8; s++) {
    r[s] = p[(size_t)s * N];
    r[s + 8] = p[(size_t)(16 + s) * N];
  }
  return r;
}

// LayerNorm over the wave's 16x128 register tile (C-fragment layout) -> f16 LDS rows.
static __device__ __forceinline__
void ln_store(const float (*nd)[8], const float* g, const float* b,
              _Float16* dst /*tile base, stride 128*/, int wv, int lane) {
  const int half = lane >> 4, l15 = lane & 15;
  float mean[8], rstd[8];
#pragma unroll
  for (int v = 0; v < 8; v++) {
    float s = 0.f;
#pragma unroll
    for (int f = 0; f < 8; f++) s += nd[f][v];
    for (int off = 8; off >= 1; off >>= 1) s += __shfl_xor(s, off, 32);
    mean[v] = s * (1.f / 128.f);
  }
#pragma unroll
  for (int v = 0; v < 8; v++) {
    float s = 0.f;
#pragma unroll
    for (int f = 0; f < 8; f++) { float d = nd[f][v] - mean[v]; s += d * d; }
    for (int off = 8; off >= 1; off >>= 1) s += __shfl_xor(s, off, 32);
    rstd[v] = rsqrtf(s * (1.f / 128.f) + 1e-5f);
  }
#pragma unroll
  for (int f = 0; f < 8; f++) {
    const int n = 16 * f + l15;
    const float gg = g[n], bb = b[n];
#pragma unroll
    for (int v = 0; v < 8; v++) {
      const int row = wv * 16 + v + 8 * half;
      dst[row * 128 + n] = (_Float16)((nd[f][v] - mean[v]) * rstd[v] * gg + bb);
    }
  }
}

// ---------------- weight f32 -> f16 convert/pack ----------------
__global__ void pack_weights(const float* __restrict__ ew1, const float* __restrict__ ew2,
                             const float* __restrict__ mw,  const float* __restrict__ rw,
                             const float* __restrict__ uw,  const float* __restrict__ cw,
                             const float* __restrict__ fw1, const float* __restrict__ fw2,
                             _Float16* __restrict__ dst) {
  long i = (long)blockIdx.x * blockDim.x + threadIdx.x;
  if (i >= TOT_H) return;
  const int l = (int)(i / LAYER_H);
  const int o = (int)(i % LAYER_H);
  float v;
  if      (o < EW2_O) v = ew1[l * 36864  + o];
  else if (o < MW_O)  v = ew2[l * 4096   + (o - EW2_O)];
  else if (o < RW_O)  v = mw [l * 4096   + (o - MW_O)];
  else if (o < UW_O)  v = rw [l * 32768  + (o - RW_O)];
  else if (o < CW_O)  v = uw [l * 32768  + (o - UW_O)];
  else if (o < FW1_O) v = cw [l * 32768  + (o - CW_O)];
  else if (o < FW2_O) v = fw1[l * 262144 + (o - FW1_O)];
  else                v = fw2[l * 262144 + (o - FW2_O)];
  dst[i] = (_Float16)v;
}

// ---------------- fused GNN kernel: one workgroup per 8-batch tile ----------------
__global__ void __launch_bounds__(TPB, 1)
rig_fused(const float* __restrict__ jf, const int* __restrict__ jtypes,
          const float* __restrict__ in_w, const float* __restrict__ in_b,
          const float* __restrict__ type_emb, const float* __restrict__ type_w,
          const float* __restrict__ type_b, const float* __restrict__ edge_dir_emb,
          const float* __restrict__ eb1, const float* __restrict__ eb2,
          const float* __restrict__ msgb, const float* __restrict__ r_bias,
          const float* __restrict__ u_bias, const float* __restrict__ c_bias,
          const float* __restrict__ fb1, const float* __restrict__ fb2,
          const float* __restrict__ n1g, const float* __restrict__ n1b,
          const float* __restrict__ n2g, const float* __restrict__ n2b,
          const float* __restrict__ outg, const float* __restrict__ outb,
          const float* __restrict__ dw, const float* __restrict__ db,
          const float* __restrict__ confw, const float* __restrict__ confb,
          const int* __restrict__ srcI, const int* __restrict__ tgtI,
          const int* __restrict__ edirI,
          const _Float16* __restrict__ pw, float* __restrict__ out) {
  extern __shared__ char smem[];
  _Float16* sNorm = (_Float16*)(smem + OFF_NORM);
  _Float16* sProj = (_Float16*)(smem + OFF_PROJ);
  _Float16* sR    = (_Float16*)(smem + OFF_R);
  _Float16* sEA   = (_Float16*)(smem + OFF_EA);
  _Float16* sAgg  = (_Float16*)(smem + OFF_AGG);
  _Float16* sScr  = (_Float16*)(smem + OFF_SCR);
  int*      sIdx  = (int*)(smem + OFF_IDX);
  float*    sRaw  = (float*)(smem + OFF_AGG);   // reused after last layer

  const int tid  = threadIdx.x;
  const int lane = tid & 31;
  const int wv   = tid >> 5;
  const int half = lane >> 4;
  const int l15  = lane & 15;
  const long gRowBase = (long)blockIdx.x * TROWS;

  if (tid < NEDGE) {
    sIdx[tid]          = srcI[tid];
    sIdx[NEDGE + tid]  = tgtI[tid];
    sIdx[2*NEDGE + tid]= edirI[tid];
  }
  __syncthreads();

  // edge_attr slot 0 init: edge_dir_emb[edir[e]]
  for (int i = tid; i < TEDGES * EDIM; i += TPB) {
    const int eL = i >> 5, d = i & 31;
    const int e = eL % NEDGE;
    sEA[i] = (_Float16)edge_dir_emb[sIdx[2*NEDGE + e] * EDIM + d];
  }

  // ---- input + type embedding -> node register tile (C-fragment layout) ----
  float node[8][8];
#pragma unroll 1
  for (int v = 0; v < 8; v++) {
    const int m = v + 8 * half;
    const long gr = gRowBase + wv * 16 + m;
    float f10[10];
    const float* jfr = jf + gr * INF;
#pragma unroll
    for (int i = 0; i < 10; i++) f10[i] = jfr[i];
    const int ty = jtypes[gr];
    float te[32];
#pragma unroll
    for (int i = 0; i < 32; i++) te[i] = type_emb[ty * 32 + i];
#pragma unroll
    for (int f = 0; f < 8; f++) {
      const int n = 16 * f + l15;
      float s = in_b[n] + type_b[n];
#pragma unroll
      for (int i = 0; i < 10; i++) s += f10[i] * in_w[i * NDIM + n];
#pragma unroll
      for (int i = 0; i < 32; i++) s += te[i] * type_w[i * NDIM + n];
      node[f][v] = s;
    }
  }

  // ---- layers ----
#pragma unroll 1
  for (int l = 0; l < NLAY; l++) {
    const size_t lb = (size_t)l * LAYER_H;
    _Float16* ea  = sEA + (l & 1) * (TEDGES * EDIM);
    _Float16* msg = sEA + ((l & 1) ^ 1) * (TEDGES * EDIM);

    // LN1 -> sNorm (f16)
    ln_store(node, n1g + l * NDIM, n1b + l * NDIM, sNorm, wv, lane);
    __syncthreads();

    // ---- edge MLP: [16,288] x [288,128] relu, then x [128,32] ----
#pragma unroll 1
    for (int et = wv; et < 21; et += NWAVE) {
      const int eL = et * 16 + l15;
      const int e = eL % NEDGE, bl = eL / NEDGE;
      const _Float16* pS = sNorm + (bl * JNT + sIdx[e]) * NDIM;
      const _Float16* pT = sNorm + (bl * JNT + sIdx[NEDGE + e]) * NDIM;
      const _Float16* pE = ea + eL * EDIM;
      f8v acc[8] = {};
#pragma unroll 1
      for (int kt = 0; kt < 9; kt++) {
        const _Float16* cb = (kt < 4) ? (pS + 32 * kt) : ((kt < 8) ? (pT + 32 * (kt - 4)) : pE);
        const h16 a = load_a(cb, half);
        h16 bf[8];
#pragma unroll
        for (int nf = 0; nf < 8; nf++) bf[nf] = load_b(pw + lb + EW1_O, 128, kt, nf, lane);
#pragma unroll
        for (int nf = 0; nf < 8; nf++) acc[nf] = wmma_f16(a, bf[nf], acc[nf]);
      }
      _Float16* scr = sScr + wv * 2048;
#pragma unroll
      for (int nf = 0; nf < 8; nf++) {
        const float bias = eb1[l * 128 + 16 * nf + l15];
#pragma unroll
        for (int v = 0; v < 8; v++) {
          float x = acc[nf][v] + bias;
          x = x > 0.f ? x : 0.f;
          scr[(v + 8 * half) * 128 + 16 * nf + l15] = (_Float16)x;
        }
      }
      f8v acc2[2] = {};
#pragma unroll 1
      for (int kt = 0; kt < 4; kt++) {
        const h16 a = load_a(scr + l15 * 128 + 32 * kt, half);
        h16 bf[2];
#pragma unroll
        for (int nf = 0; nf < 2; nf++) bf[nf] = load_b(pw + lb + EW2_O, 32, kt, nf, lane);
#pragma unroll
        for (int nf = 0; nf < 2; nf++) acc2[nf] = wmma_f16(a, bf[nf], acc2[nf]);
      }
#pragma unroll
      for (int nf = 0; nf < 2; nf++) {
        const float bias = eb2[l * 32 + 16 * nf + l15];
#pragma unroll
        for (int v = 0; v < 8; v++)
          msg[(et * 16 + v + 8 * half) * EDIM + 16 * nf + l15] = (_Float16)(acc2[nf][v] + bias);
      }
    }
    __syncthreads();

    // ---- aggregate messages per target node (tree, 42 edges) ----
#pragma unroll 1
    for (int i = tid; i < TROWS * EDIM; i += TPB) {
      const int row = i >> 5, d = i & 31;
      const int bl = row / JNT, j = row % JNT;
      float s = 0.f;
#pragma unroll 1
      for (int e = 0; e < NEDGE; e++)
        if (sIdx[NEDGE + e] == j) s += (float)msg[(bl * NEDGE + e) * EDIM + d];
      sAgg[row * EDIM + d] = (_Float16)s;
    }
    __syncthreads();

    // ---- proj = agg32 @ msg_w + msg_b ----
    {
      f8v acc[8] = {};
      const h16 a = load_a(sAgg + (wv * 16 + l15) * EDIM, half);
      h16 bf[8];
#pragma unroll
      for (int nf = 0; nf < 8; nf++) bf[nf] = load_b(pw + lb + MW_O, 128, 0, nf, lane);
#pragma unroll
      for (int nf = 0; nf < 8; nf++) acc[nf] = wmma_f16(a, bf[nf], acc[nf]);
#pragma unroll
      for (int nf = 0; nf < 8; nf++) {
        const float bias = msgb[l * 128 + 16 * nf + l15];
#pragma unroll
        for (int v = 0; v < 8; v++)
          sProj[(wv * 16 + v + 8 * half) * NDIM + 16 * nf + l15] = (_Float16)(acc[nf][v] + bias);
      }
    }

    // ---- GRU gates (wave-local rows, no barrier needed) ----
    const _Float16* myN = sNorm + (wv * 16 + l15) * NDIM;
    const _Float16* myP = sProj + (wv * 16 + l15) * NDIM;
    _Float16* myU = sScr + wv * 2048;
    {   // r gate
      f8v aR[8] = {};
#pragma unroll 1
      for (int kt = 0; kt < 8; kt++) {
        const _Float16* cb = (kt < 4) ? (myN + 32 * kt) : (myP + 32 * (kt - 4));
        const h16 a = load_a(cb, half);
        h16 bf[8];
#pragma unroll
        for (int nf = 0; nf < 8; nf++) bf[nf] = load_b(pw + lb + RW_O, 128, kt, nf, lane);
#pragma unroll
        for (int nf = 0; nf < 8; nf++) aR[nf] = wmma_f16(a, bf[nf], aR[nf]);
      }
#pragma unroll
      for (int nf = 0; nf < 8; nf++) {
        const float bias = r_bias[l * 128 + 16 * nf + l15];
#pragma unroll
        for (int v = 0; v < 8; v++) {
          const float rv = 1.f / (1.f + expf(-(aR[nf][v] + bias)));
          sR[(wv * 16 + v + 8 * half) * NDIM + 16 * nf + l15] = (_Float16)rv;
        }
      }
    }
    {   // u gate -> per-wave scratch
      f8v aU[8] = {};
#pragma unroll 1
      for (int kt = 0; kt < 8; kt++) {
        const _Float16* cb = (kt < 4) ? (myN + 32 * kt) : (myP + 32 * (kt - 4));
        const h16 a = load_a(cb, half);
        h16 bf[8];
#pragma unroll
        for (int nf = 0; nf < 8; nf++) bf[nf] = load_b(pw + lb + UW_O, 128, kt, nf, lane);
#pragma unroll
        for (int nf = 0; nf < 8; nf++) aU[nf] = wmma_f16(a, bf[nf], aU[nf]);
      }
#pragma unroll
      for (int nf = 0; nf < 8; nf++) {
        const float bias = u_bias[l * 128 + 16 * nf + l15];
#pragma unroll
        for (int v = 0; v < 8; v++) {
          const float uv = 1.f / (1.f + expf(-(aU[nf][v] + bias)));
          myU[(v + 8 * half) * 128 + 16 * nf + l15] = (_Float16)uv;
        }
      }
    }
    // rn = r * normed (overwrite sR rows of this wave; rows are contiguous)
#pragma unroll 1
    for (int i = lane; i < 16 * 128; i += 32)
      sR[wv * 2048 + i] = (_Float16)((float)sR[wv * 2048 + i] * (float)sNorm[wv * 2048 + i]);
    // cand = tanh([rn, proj] @ c_w + c_b); update node regs
    {
      f8v aC[8] = {};
      const _Float16* myRN = sR + (wv * 16 + l15) * NDIM;
#pragma unroll 1
      for (int kt = 0; kt < 8; kt++) {
        const _Float16* cb = (kt < 4) ? (myRN + 32 * kt) : (myP + 32 * (kt - 4));
        const h16 a = load_a(cb, half);
        h16 bf[8];
#pragma unroll
        for (int nf = 0; nf < 8; nf++) bf[nf] = load_b(pw + lb + CW_O, 128, kt, nf, lane);
#pragma unroll
        for (int nf = 0; nf < 8; nf++) aC[nf] = wmma_f16(a, bf[nf], aC[nf]);
      }
#pragma unroll
      for (int nf = 0; nf < 8; nf++) {
        const float bias = c_bias[l * 128 + 16 * nf + l15];
#pragma unroll
        for (int v = 0; v < 8; v++) {
          const float cand = tanhf(aC[nf][v] + bias);
          const int li = (v + 8 * half) * 128 + 16 * nf + l15;
          const float nm = (float)sNorm[wv * 2048 + li];
          const float uu = (float)myU[li];
          node[nf][v] += (1.f - uu) * nm + uu * cand;
        }
      }
    }

    // ---- FFN: stream hidden dim in chunks of 32 ----
    ln_store(node, n2g + l * NDIM, n2b + l * NDIM, sNorm, wv, lane);
    {
      f8v oAcc[8] = {};
      const _Float16* myN2 = sNorm + (wv * 16 + l15) * NDIM;
      _Float16* scrF = sScr + wv * 2048;   // used as [16][32]
      // A fragments of normed2 are invariant across all 64 chunks: hoist.
      h16 aN2[4];
#pragma unroll
      for (int kt = 0; kt < 4; kt++) aN2[kt] = load_a(myN2 + 32 * kt, half);
#pragma unroll 1
      for (int c = 0; c < 64; c++) {
        h16 b1f[8];
#pragma unroll
        for (int kt = 0; kt < 4; kt++)
#pragma unroll
          for (int nf = 0; nf < 2; nf++)
            b1f[kt * 2 + nf] = load_b(pw + lb + FW1_O, 2048, kt, 2 * c + nf, lane);
        f8v a1[2] = {};
#pragma unroll
        for (int kt = 0; kt < 4; kt++)
#pragma unroll
          for (int nf = 0; nf < 2; nf++)
            a1[nf] = wmma_f16(aN2[kt], b1f[kt * 2 + nf], a1[nf]);
#pragma unroll
        for (int nf = 0; nf < 2; nf++) {
          const float bias = fb1[l * FFND + 32 * c + 16 * nf + l15];
#pragma unroll
          for (int v = 0; v < 8; v++) {
            float x = a1[nf][v] + bias;
            x = 0.5f * x * (1.0f + erff(x * 0.70710678118654752f));   // exact GELU
            scrF[(v + 8 * half) * 32 + 16 * nf + l15] = (_Float16)x;
          }
        }
        const h16 a2 = load_a(scrF + l15 * 32, half);
        h16 b2f[8];
#pragma unroll
        for (int nf = 0; nf < 8; nf++) b2f[nf] = load_b(pw + lb + FW2_O, 128, c, nf, lane);
#pragma unroll
        for (int nf = 0; nf < 8; nf++) oAcc[nf] = wmma_f16(a2, b2f[nf], oAcc[nf]);
      }
#pragma unroll
      for (int nf = 0; nf < 8; nf++) {
        const float bias = fb2[l * 128 + 16 * nf + l15];
#pragma unroll
        for (int v = 0; v < 8; v++) node[nf][v] += oAcc[nf][v] + bias;
      }
    }
    __syncthreads();
  }

  // ---- output heads ----
  ln_store(node, outg, outb, sNorm, wv, lane);
  __syncthreads();
#pragma unroll 1
  for (int i = tid; i < TROWS * 4; i += TPB) {
    const int row = i >> 2, d = i & 3;
    float s = db[d];
    const _Float16* r_ = sNorm + row * NDIM;
#pragma unroll 1
    for (int n = 0; n < NDIM; n++) s += (float)r_[n] * dw[n * 4 + d];
    sRaw[row * 4 + d] = s;
  }
  __syncthreads();
#pragma unroll 1
  for (int row = tid; row < TROWS; row += TPB) {
    const float r0 = sRaw[row * 4 + 0], r1 = sRaw[row * 4 + 1];
    const float r2 = sRaw[row * 4 + 2], r3 = sRaw[row * 4 + 3];
    const float nrm = fmaxf(sqrtf(r0 * r0 + r1 * r1 + r2 * r2 + r3 * r3), 1e-8f);
    const long gr = gRowBase + row;
    out[gr * 4 + 0] = r0 / nrm;
    out[gr * 4 + 1] = r1 / nrm;
    out[gr * 4 + 2] = r2 / nrm;
    out[gr * 4 + 3] = r3 / nrm;
    float s = confb[0];
    const _Float16* r_ = sNorm + row * NDIM;
#pragma unroll 1
    for (int n = 0; n < NDIM; n++) s += (float)r_[n] * confw[n];
    out[(long)BATCH * JNT * 4 + gr] = 1.f / (1.f + expf(-s));
  }
}

// ---------------- host launch ----------------
extern "C" void kernel_launch(void* const* d_in, const int* in_sizes, int n_in,
                              void* d_out, int out_size, void* d_ws, size_t ws_size,
                              hipStream_t stream) {
  (void)in_sizes; (void)n_in; (void)out_size; (void)ws_size;
  // setup_inputs() insertion order, params flattened recursively in insertion order
  const float* jf        = (const float*)d_in[0];
  const int*   jtypes    = (const int*)  d_in[1];
  const float* in_w      = (const float*)d_in[2];
  const float* in_b      = (const float*)d_in[3];
  const float* type_emb  = (const float*)d_in[4];
  const float* type_w    = (const float*)d_in[5];
  const float* type_b    = (const float*)d_in[6];
  const float* edge_dire = (const float*)d_in[7];
  const float* edge_w1   = (const float*)d_in[8];
  const float* edge_b1   = (const float*)d_in[9];
  const float* edge_w2   = (const float*)d_in[10];
  const float* edge_b2   = (const float*)d_in[11];
  const float* msg_w     = (const float*)d_in[12];
  const float* msg_b     = (const float*)d_in[13];
  const float* r_w       = (const float*)d_in[14];
  const float* r_b       = (const float*)d_in[15];
  const float* u_w       = (const float*)d_in[16];
  const float* u_b       = (const float*)d_in[17];
  const float* c_w       = (const float*)d_in[18];
  const float* c_b       = (const float*)d_in[19];
  const float* ffn_w1    = (const float*)d_in[20];
  const float* ffn_b1    = (const float*)d_in[21];
  const float* ffn_w2    = (const float*)d_in[22];
  const float* ffn_b2    = (const float*)d_in[23];
  const float* n1_g      = (const float*)d_in[24];
  const float* n1_b      = (const float*)d_in[25];
  const float* n2_g      = (const float*)d_in[26];
  const float* n2_b      = (const float*)d_in[27];
  const float* out_g     = (const float*)d_in[28];
  const float* out_b     = (const float*)d_in[29];
  const float* delta_w   = (const float*)d_in[30];
  const float* delta_b   = (const float*)d_in[31];
  const float* conf_w    = (const float*)d_in[32];
  const float* conf_b    = (const float*)d_in[33];
  const int*   src_idx   = (const int*)  d_in[34];
  const int*   tgt_idx   = (const int*)  d_in[35];
  /* d_in[36] = agg_mat (unused; tree aggregation derived from tgt_idx) */
  const int*   edge_dir  = (const int*)  d_in[37];

  _Float16* pw = (_Float16*)d_ws;
  float* outp = (float*)d_out;

  // 1) convert all GEMM weights to f16 (L2-resident, read as WMMA B fragments)
  {
    const int thr = 256;
    const int blk = (int)((TOT_H + thr - 1) / thr);
    pack_weights<<<blk, thr, 0, stream>>>(edge_w1, edge_w2, msg_w, r_w, u_w, c_w,
                                          ffn_w1, ffn_w2, pw);
  }

  // 2) fused persistent-tile GNN kernel
  (void)hipFuncSetAttribute((const void*)rig_fused,
                            hipFuncAttributeMaxDynamicSharedMemorySize, SMEM_BYTES);
  rig_fused<<<BATCH / TBATCH, TPB, SMEM_BYTES, stream>>>(
      jf, jtypes, in_w, in_b, type_emb, type_w, type_b, edge_dire,
      edge_b1, edge_b2, msg_b, r_b, u_b, c_b, ffn_b1, ffn_b2,
      n1_g, n1_b, n2_g, n2_b, out_g, out_b, delta_w, delta_b, conf_w, conf_b,
      src_idx, tgt_idx, edge_dir, pw, outp);
}
